// FastAttention_78142634983568
// MI455X (gfx1250) — matrix operand order repaired and checked by reading, not compiled, not run on hardware
//
#include <hip/hip_runtime.h>
#include <hip/hip_bf16.h>

#define N_NODES 65536
#define FDIM 512
#define BR 32           // Q/K rows per block in phi kernels
#define XLD 520         // padded LDS row stride (bank-conflict-free A frags)
#define KV_BM 64        // kv kernel: feature tile
#define KV_BD 64        // kv kernel: dv tile
#define KV_CH 64        // kv kernel: n-chunk per TDM stage
#define KV_LDP 68       // kv kernel padded LDS stride
#define KV_SPLIT 8      // split-K over n

typedef __attribute__((ext_vector_type(16))) __bf16 v16bf;
typedef __attribute__((ext_vector_type(8)))  float  v8f;
typedef __attribute__((ext_vector_type(4)))  unsigned int v4u;
typedef __attribute__((ext_vector_type(8)))  int v8i;
typedef __attribute__((ext_vector_type(4)))  int v4i;

union FragBF { v16bf v; unsigned int u[8]; unsigned short h[16]; };

__device__ __forceinline__ unsigned short f2bf(float f) {
  union { float f; unsigned u; } x; x.f = f;
  unsigned r = x.u + 0x7FFFu + ((x.u >> 16) & 1u);
  return (unsigned short)(r >> 16);
}
__device__ __forceinline__ float bf2f(unsigned short b) {
  union { unsigned u; float f; } x; x.u = ((unsigned)b) << 16;
  return x.f;
}

__device__ __forceinline__ v8f wmma_bf16(v16bf a, v16bf b, v8f c) {
  return __builtin_amdgcn_wmma_f32_16x16x32_bf16(false, a, false, b, (short)0, c,
                                                 false, false);
}

// ---------------------------------------------------------------------------
// TDM: DMA a [tile_h x tile_w] tile out of a row-major [65536 x 512] tensor
// (row stride 512 elements) into LDS. dsz_code: 1 = 2B, 2 = 4B elements.
__device__ __forceinline__ void tdm_load_2d(const void* gsrc, void* lds_dst,
                                            int dsz_code, int tile_w, int tile_h) {
  unsigned long long ga = (unsigned long long)(uintptr_t)gsrc;
  v4u g0;
  g0.x = 1u;                                       // count = 1 valid descriptor
  g0.y = (unsigned int)(uintptr_t)lds_dst;         // LDS byte address (addr[31:0])
  g0.z = (unsigned int)ga;                         // global_addr lo
  g0.w = ((unsigned int)(ga >> 32) & 0x01FFFFFFu)  // global_addr hi (bits 56:32)
         | 0x80000000u;                            // type = 2 ("image")
  v8i g1;
  g1[0] = dsz_code << 16;            // data_size, no flags
  g1[1] = (int)(512u << 16);         // tensor_dim0 = 512 (bits 63:48)
  g1[2] = 0;                         // tensor_dim0 hi, tensor_dim1 lo16 (65536)
  g1[3] = 1 | (tile_w << 16);        // tensor_dim1 hi = 1, tile_dim0
  g1[4] = tile_h;                    // tile_dim1, tile_dim2 = 0
  g1[5] = 512;                       // tensor_dim0_stride = 512
  g1[6] = 0;
  g1[7] = 0;
  v4i z4 = {0, 0, 0, 0};
#if defined(__clang_major__) && (__clang_major__ >= 23)
  v8i z8 = {0, 0, 0, 0, 0, 0, 0, 0};
  __builtin_amdgcn_tensor_load_to_lds(g0, g1, z4, z4, z8, 0);
#else
  __builtin_amdgcn_tensor_load_to_lds(g0, g1, z4, z4, 0);
#endif
}

// A fragment (16x32 bf16) from a row-major [16 x ld] bf16 tile at t, k-offset k0.
__device__ __forceinline__ v16bf load_a_rowmajor(const unsigned short* t, int ld,
                                                 int k0, int lane) {
  FragBF fa;
  int m = lane & 15, half = lane >> 4;
  int kb = k0 + half * 8;
  const unsigned short* row = t + m * ld;
#pragma unroll
  for (int v = 0; v < 4; ++v)
    fa.u[v] = *(const unsigned int*)(row + kb + 2 * v);
#pragma unroll
  for (int v = 0; v < 4; ++v)
    fa.u[4 + v] = *(const unsigned int*)(row + kb + 16 + 2 * v);
  return fa.v;
}

// B fragment (32x16 bf16) from a column-major tile stored as [16 cols x ld].
__device__ __forceinline__ v16bf load_b_colmajor(const unsigned short* t, int ld,
                                                 int k0, int lane) {
  FragBF fb;
  int n = lane & 15, half = lane >> 4;
  int kb = k0 + half * 8;
  const unsigned short* col = t + n * ld;
#pragma unroll
  for (int v = 0; v < 4; ++v)
    fb.u[v] = *(const unsigned int*)(col + kb + 2 * v);
#pragma unroll
  for (int v = 0; v < 4; ++v)
    fb.u[4 + v] = *(const unsigned int*)(col + kb + 16 + 2 * v);
  return fb.v;
}

// ---------------------------------------------------------------------------
// Kernel 1: omega -> bf16, zero kv / k_ones accumulators.
__global__ __launch_bounds__(256) void prep_kernel(
    const float* __restrict__ omega, unsigned short* __restrict__ om_bf,
    float* __restrict__ kv, float* __restrict__ k_ones) {
  int idx = blockIdx.x * 256 + threadIdx.x;
  for (int i = idx; i < FDIM * FDIM; i += 512 * 256) {
    om_bf[i] = f2bf(omega[i]);
    kv[i] = 0.f;
  }
  if (idx < FDIM) k_ones[idx] = 0.f;
}

// ---------------------------------------------------------------------------
// Kernel 2: K_hat = phi(K) in bf16. TDM stages the 32x512 fp32 K tile to LDS;
// each wave shares one omega B-fragment across a pair of row-tiles.
__global__ __launch_bounds__(256) void phi_k_kernel(
    const float* __restrict__ K, const unsigned short* __restrict__ om_bf,
    unsigned short* __restrict__ Khat) {
  __shared__ __align__(16) float stagef[BR * FDIM];   // 64 KB fp32 via TDM
  __shared__ unsigned short xt[BR * XLD];             // ~33 KB bf16 (padded)
  __shared__ float part[BR][8];
  __shared__ float hrow[BR];
  int t = threadIdx.x;
  int row0 = blockIdx.x * BR;

  __builtin_prefetch(om_bf + ((size_t)t << 10), 0, 3);

  if (t < 32) {  // wave 0 issues the TDM descriptor and waits on TENSORcnt
    tdm_load_2d(K + (size_t)row0 * FDIM, (void*)stagef, 2, 512, BR);
    __builtin_amdgcn_s_wait_tensorcnt(0);
  }
  __syncthreads();

  // fp32 LDS -> padded bf16 LDS
#pragma unroll
  for (int i = 0; i < (BR * FDIM) / 256; ++i) {
    int idx = i * 256 + t;
    int r = idx >> 9, c = idx & 511;
    xt[r * XLD + c] = f2bf(stagef[idx]);
  }
  __syncthreads();
  {  // per-row ||x||^2 (fp32) -> h = exp(-0.5*s)/sqrt(512)
    int r = t >> 3, p = t & 7;
    float s = 0.f;
    for (int c = p * 64; c < p * 64 + 64; ++c) {
      float x = stagef[r * FDIM + c];
      s += x * x;
    }
    part[r][p] = s;
  }
  __syncthreads();
  if (t < BR) {
    float s = 0.f;
#pragma unroll
    for (int p = 0; p < 8; ++p) s += part[t][p];
    hrow[t] = __expf(-0.5f * s) * 0.044194173824159216f;  // 1/sqrt(512)
  }
  __syncthreads();

  int w = t >> 5, lane = t & 31;
  for (int ct = w; ct < 32; ct += 8) {  // col-tile; both row-tiles share B
    int col0 = ct * 16;
    v8f acc0 = {}, acc1 = {};
#pragma unroll 4
    for (int k0 = 0; k0 < FDIM; k0 += 32) {
      v16bf b = load_b_colmajor(om_bf + (size_t)col0 * FDIM, FDIM, k0, lane);
      v16bf a0 = load_a_rowmajor(xt, XLD, k0, lane);
      v16bf a1 = load_a_rowmajor(xt + 16 * XLD, XLD, k0, lane);
      acc0 = wmma_bf16(a0, b, acc0);
      acc1 = wmma_bf16(a1, b, acc1);
    }
    int n = lane & 15, half = lane >> 4;
#pragma unroll
    for (int v = 0; v < 8; ++v) {
      int m = v + 8 * half;
      Khat[(size_t)(row0 + m) * FDIM + col0 + n] =
          f2bf(hrow[m] * __expf(acc0[v]));
      Khat[(size_t)(row0 + 16 + m) * FDIM + col0 + n] =
          f2bf(hrow[16 + m] * __expf(acc1[v]));
    }
  }
}

// ---------------------------------------------------------------------------
// Kernel 3: kv = K_hat^T @ V, TDM double-buffered over 64-row n-chunks with
// split-K atomics; dv-tile-0 blocks also accumulate k_ones = colsum(K_hat).
__global__ __launch_bounds__(256) void kv_kernel(
    const unsigned short* __restrict__ Khat, const float* __restrict__ V,
    float* __restrict__ kv, float* __restrict__ k_ones) {
  __shared__ unsigned short kraw[2][KV_CH * KV_BM];       // 2 x 8 KB (TDM dst)
  __shared__ __align__(16) float vraw[2][KV_CH * KV_BD];  // 2 x 16 KB (TDM dst)
  __shared__ unsigned short ldsA[KV_BM * KV_LDP];         // [f][n] padded
  __shared__ unsigned short ldsB[KV_BD * KV_LDP];         // [dv][n] padded
  int t = threadIdx.x, w = t >> 5, lane = t & 31;
  int f0 = blockIdx.x * KV_BM;
  int d0 = blockIdx.y * KV_BD;
  int nbeg = blockIdx.z * (N_NODES / KV_SPLIT);
  const int nch = (N_NODES / KV_SPLIT) / KV_CH;

  int ti0 = w, ti1 = w + 8;  // each wave owns 2 of the 16 16x16 tiles
  int rt0 = ti0 >> 2, ct0 = ti0 & 3;
  int rt1 = ti1 >> 2, ct1 = ti1 & 3;
  v8f acc0 = {}, acc1 = {};
  float kones_acc = 0.f;
  bool do_kones = (blockIdx.y == 0) && (t < KV_BM);

  if (t < 32) {  // prime the TDM pipeline with chunk 0
    tdm_load_2d(Khat + (size_t)nbeg * FDIM + f0, kraw[0], 1, KV_BM, KV_CH);
    tdm_load_2d(V + (size_t)nbeg * FDIM + d0, vraw[0], 2, KV_BD, KV_CH);
  }
  int buf = 0;
  for (int i = 0; i < nch; ++i) {
    if (t < 32) {
      if (i + 1 < nch) {  // overlap: DMA chunk i+1 while computing chunk i
        size_t nn = (size_t)(nbeg + (i + 1) * KV_CH);
        tdm_load_2d(Khat + nn * FDIM + f0, kraw[buf ^ 1], 1, KV_BM, KV_CH);
        tdm_load_2d(V + nn * FDIM + d0, vraw[buf ^ 1], 2, KV_BD, KV_CH);
        __builtin_amdgcn_s_wait_tensorcnt(2);  // in-order: chunk i landed
      } else {
        __builtin_amdgcn_s_wait_tensorcnt(0);
      }
    }
    __syncthreads();
    // transpose raw [n][e] -> padded compute tiles [e][n] (V: fp32->bf16)
#pragma unroll
    for (int j = 0; j < (KV_CH * KV_BM) / 256; ++j) {
      int idx = j * 256 + t;
      int nn = idx >> 6, ee = idx & 63;
      ldsA[ee * KV_LDP + nn] = kraw[buf][idx];
      ldsB[ee * KV_LDP + nn] = f2bf(vraw[buf][idx]);
    }
    __syncthreads();
    if (do_kones) {
      float s = 0.f;
#pragma unroll
      for (int nn = 0; nn < KV_CH; ++nn) s += bf2f(ldsA[t * KV_LDP + nn]);
      kones_acc += s;
    }
#pragma unroll
    for (int k0 = 0; k0 < KV_CH; k0 += 32) {
      v16bf a0 = load_a_rowmajor(ldsA + rt0 * 16 * KV_LDP, KV_LDP, k0, lane);
      v16bf b0 = load_b_colmajor(ldsB + ct0 * 16 * KV_LDP, KV_LDP, k0, lane);
      acc0 = wmma_bf16(a0, b0, acc0);
      v16bf a1 = load_a_rowmajor(ldsA + rt1 * 16 * KV_LDP, KV_LDP, k0, lane);
      v16bf b1 = load_b_colmajor(ldsB + ct1 * 16 * KV_LDP, KV_LDP, k0, lane);
      acc1 = wmma_bf16(a1, b1, acc1);
    }
    buf ^= 1;
    __syncthreads();  // ldsA/ldsB stable until all waves finished fragments
  }

  int n = lane & 15, half = lane >> 4;
#pragma unroll
  for (int v = 0; v < 8; ++v) {
    int m = v + 8 * half;
    atomicAdd(&kv[(size_t)(f0 + rt0 * 16 + m) * FDIM + d0 + ct0 * 16 + n], acc0[v]);
    atomicAdd(&kv[(size_t)(f0 + rt1 * 16 + m) * FDIM + d0 + ct1 * 16 + n], acc1[v]);
  }
  if (do_kones) atomicAdd(&k_ones[f0 + t], kones_acc);
}

// ---------------------------------------------------------------------------
// Kernel 3b: kvT (bf16, column-major of kv) so qkv B-fragments are contiguous.
__global__ __launch_bounds__(256) void kvt_kernel(
    const float* __restrict__ kv, unsigned short* __restrict__ kvT) {
  int idx = blockIdx.x * 256 + threadIdx.x;  // 1024 blocks x 256 = 512*512
  int m = idx >> 9, dv = idx & 511;
  kvT[dv * FDIM + m] = f2bf(kv[idx]);
}

// ---------------------------------------------------------------------------
// Kernel 4: out = (phi(Q) @ kv) / (phi(Q) @ k_ones). TDM stages Q; the 64 KB
// fp32 staging buffer is reused as the Q_hat bf16 buffer after conversion.
__global__ __launch_bounds__(256) void qkv_kernel(
    const float* __restrict__ Q, const unsigned short* __restrict__ om_bf,
    const unsigned short* __restrict__ kvT, const float* __restrict__ k_ones,
    float* __restrict__ out) {
  __shared__ __align__(16) char smem[BR * FDIM * 4];  // 64 KB: stagef then qhat
  __shared__ unsigned short xt[BR * XLD];             // ~33 KB staged Q (bf16)
  __shared__ float part[BR][8];
  __shared__ float hrow[BR];
  __shared__ float dinv[BR];
  float* stagef = (float*)smem;
  unsigned short* qhat = (unsigned short*)smem;       // stride XLD, ~33 KB
  int t = threadIdx.x;
  int row0 = blockIdx.x * BR;

  __builtin_prefetch(om_bf + ((size_t)t << 10), 0, 3);
  __builtin_prefetch(kvT + ((size_t)t << 10), 0, 3);

  if (t < 32) {
    tdm_load_2d(Q + (size_t)row0 * FDIM, (void*)stagef, 2, 512, BR);
    __builtin_amdgcn_s_wait_tensorcnt(0);
  }
  __syncthreads();

#pragma unroll
  for (int i = 0; i < (BR * FDIM) / 256; ++i) {
    int idx = i * 256 + t;
    int r = idx >> 9, c = idx & 511;
    xt[r * XLD + c] = f2bf(stagef[idx]);
  }
  __syncthreads();
  {
    int r = t >> 3, p = t & 7;
    float s = 0.f;
    for (int c = p * 64; c < p * 64 + 64; ++c) {
      float x = stagef[r * FDIM + c];
      s += x * x;
    }
    part[r][p] = s;
  }
  __syncthreads();
  if (t < BR) {
    float s = 0.f;
#pragma unroll
    for (int p = 0; p < 8; ++p) s += part[t][p];
    hrow[t] = __expf(-0.5f * s) * 0.044194173824159216f;
  }
  __syncthreads();  // stagef dead beyond this point -> safe to write qhat

  int w = t >> 5, lane = t & 31;
  // Phase 1: Q_hat into LDS (WMMA vs omega + exp epilogue), B shared per pair
  for (int ct = w; ct < 32; ct += 8) {
    int col0 = ct * 16;
    v8f acc0 = {}, acc1 = {};
#pragma unroll 4
    for (int k0 = 0; k0 < FDIM; k0 += 32) {
      v16bf b = load_b_colmajor(om_bf + (size_t)col0 * FDIM, FDIM, k0, lane);
      v16bf a0 = load_a_rowmajor(xt, XLD, k0, lane);
      v16bf a1 = load_a_rowmajor(xt + 16 * XLD, XLD, k0, lane);
      acc0 = wmma_bf16(a0, b, acc0);
      acc1 = wmma_bf16(a1, b, acc1);
    }
    int n = lane & 15, half = lane >> 4;
#pragma unroll
    for (int v = 0; v < 8; ++v) {
      int m = v + 8 * half;
      qhat[m * XLD + col0 + n] = f2bf(hrow[m] * __expf(acc0[v]));
      qhat[(16 + m) * XLD + col0 + n] = f2bf(hrow[16 + m] * __expf(acc1[v]));
    }
  }
  __syncthreads();

  // d_diag = Q_hat @ k_ones (per-row)
  {
    int r = t >> 3, p = t & 7;
    float s = 0.f;
    for (int c = p * 64; c < p * 64 + 64; ++c)
      s += bf2f(qhat[r * XLD + c]) * k_ones[c];
    part[r][p] = s;
  }
  __syncthreads();
  if (t < BR) {
    float s = 0.f;
#pragma unroll
    for (int p = 0; p < 8; ++p) s += part[t][p];
    dinv[t] = 1.0f / s;
  }
  __syncthreads();

  // Phase 2: out = (Q_hat @ kv) * dinv, B from bf16 col-major kvT (L2)
  for (int ct = w; ct < 32; ct += 8) {
    int col0 = ct * 16;
    v8f acc0 = {}, acc1 = {};
#pragma unroll 4
    for (int k0 = 0; k0 < FDIM; k0 += 32) {
      v16bf b = load_b_colmajor(kvT + (size_t)col0 * FDIM, FDIM, k0, lane);
      v16bf a0 = load_a_rowmajor(qhat, XLD, k0, lane);
      v16bf a1 = load_a_rowmajor(qhat + 16 * XLD, XLD, k0, lane);
      acc0 = wmma_bf16(a0, b, acc0);
      acc1 = wmma_bf16(a1, b, acc1);
    }
    int n = lane & 15, half = lane >> 4;
#pragma unroll
    for (int v = 0; v < 8; ++v) {
      int m = v + 8 * half;
      out[(size_t)(row0 + m) * FDIM + col0 + n] = acc0[v] * dinv[m];
      out[(size_t)(row0 + 16 + m) * FDIM + col0 + n] = acc1[v] * dinv[16 + m];
    }
  }
}

// ---------------------------------------------------------------------------
extern "C" void kernel_launch(void* const* d_in, const int* in_sizes, int n_in,
                              void* d_out, int out_size, void* d_ws, size_t ws_size,
                              hipStream_t stream) {
  const float* Q = (const float*)d_in[0];
  const float* K = (const float*)d_in[1];
  const float* V = (const float*)d_in[2];
  const float* omega = (const float*)d_in[3];
  float* out = (float*)d_out;

  char* ws = (char*)d_ws;
  unsigned short* Khat = (unsigned short*)ws;                       // 64 MB
  size_t off = (size_t)N_NODES * FDIM * sizeof(unsigned short);
  unsigned short* om_bf = (unsigned short*)(ws + off);              // 512 KB
  off += (size_t)FDIM * FDIM * sizeof(unsigned short);
  float* kv = (float*)(ws + off);                                   // 1 MB
  off += (size_t)FDIM * FDIM * sizeof(float);
  unsigned short* kvT = (unsigned short*)(ws + off);                // 512 KB
  off += (size_t)FDIM * FDIM * sizeof(unsigned short);
  float* k_ones = (float*)(ws + off);                               // 2 KB

  prep_kernel<<<512, 256, 0, stream>>>(omega, om_bf, kv, k_ones);
  phi_k_kernel<<<N_NODES / BR, 256, 0, stream>>>(K, om_bf, Khat);
  kv_kernel<<<dim3(FDIM / KV_BM, FDIM / KV_BD, KV_SPLIT), 256, 0, stream>>>(
      Khat, V, kv, k_ones);
  kvt_kernel<<<(FDIM * FDIM) / 256, 256, 0, stream>>>(kv, kvT);
  qkv_kernel<<<N_NODES / BR, 256, 0, stream>>>(Q, om_bf, kvT, k_ones, out);
}
